// SelfAttention_4398046511910
// MI455X (gfx1250) — matrix operand order repaired and checked by reading, not compiled, not run on hardware
//
#include <hip/hip_runtime.h>
#include <hip/hip_bf16.h>

// ---------------------------------------------------------------------------
// SelfAttention (attention-over-heads variant), MI455X / gfx1250, wave32.
//   qkv = x @ W_attn + b_attn          (bf16 WMMA GEMM, LDS-staged B, f32 acc)
//   S   = Q Kt / 8  (16x16 per token)  (v_wmma_f32_16x16x32_bf16 x2)
//   P   = softmax(S, axis=g)           (lane butterflies)
//   O   = P V                          (v_wmma x4, K padded 16->32)
//   out = O @ W_proj + b_proj          (same GEMM, f32 out)
// WMMA layout contracts (CDNA5 ISA):
//   A 16-bit frag: lane m = L%16, elem e -> K = (e/8)*16 + (L/16)*8 + e%8
//   B 16-bit frag: lane n = L%16, elem e -> K = (e/8)*16 + (L/16)*8 + e%8
//   C/D f32: VGPR r, lane L -> M = r + 8*(L/16), N = L%16
// ---------------------------------------------------------------------------

typedef __attribute__((ext_vector_type(16))) __bf16 v16bf;
typedef __attribute__((ext_vector_type(8)))  __bf16 v8bf;
typedef __attribute__((ext_vector_type(8)))  float  v8f;

// gcc-style int4 vector with explicit address spaces, matching the async
// global->LDS builtin's parameter types (per hipcc diagnostic: param1 is
// "int __vector(4) __device__ *", i.e. v4i addrspace(1)*).
typedef int v4i __attribute__((__vector_size__(4 * sizeof(int))));
typedef __attribute__((address_space(1))) v4i gv4i;
typedef __attribute__((address_space(3))) v4i lv4i;

#define B_SZ   8
#define T_SZ   2048
#define C_SZ   1024
#define H_SZ   16
#define HD_SZ  64
#define M_ROWS 16384   /* B*T */
#define N1_SZ  3072
#define N2_SZ  1024
#define K_SZ   1024

// gfx1250 async global->LDS path (probe-guarded; falls back to sync loads)
#if defined(__has_builtin)
#  if __has_builtin(__builtin_amdgcn_global_load_async_to_lds_b128) && \
      __has_builtin(__builtin_amdgcn_s_wait_asynccnt)
#    define USE_ASYNC_LDS 1
#  endif
#endif
#ifndef USE_ASYNC_LDS
#  define USE_ASYNC_LDS 0
#endif

static __device__ __forceinline__ unsigned short bf16_bits(float f) {
  unsigned int u = __builtin_bit_cast(unsigned int, f);
  unsigned int r = u + 0x7FFFu + ((u >> 16) & 1u);   // round-to-nearest-even
  return (unsigned short)(r >> 16);
}
static __device__ __forceinline__ __bf16 f2bf(float f) {
  unsigned short s = bf16_bits(f);
  return __builtin_bit_cast(__bf16, s);
}

// ---------------- x (fp32) -> bf16, vectorized ----------------
__global__ void k_cvt_bf16(const float* __restrict__ src,
                           __bf16* __restrict__ dst, int n4) {
  int i = blockIdx.x * blockDim.x + threadIdx.x;
  int stride = gridDim.x * blockDim.x;
  const float4* s4 = (const float4*)src;
  uint2* d2 = (uint2*)dst;
  for (; i < n4; i += stride) {
    float4 v = s4[i];
    unsigned int lo = (unsigned int)bf16_bits(v.x) | ((unsigned int)bf16_bits(v.y) << 16);
    unsigned int hi = (unsigned int)bf16_bits(v.z) | ((unsigned int)bf16_bits(v.w) << 16);
    d2[i] = make_uint2(lo, hi);
  }
}

// ---------------- W (R x Cc, fp32) -> Wt (Cc x R, bf16), LDS tiled ----------
__global__ void k_transpose_to_bf16(const float* __restrict__ src,
                                    __bf16* __restrict__ dst, int R, int Cc) {
  __shared__ float tile[16][17];
  int tx = threadIdx.x & 15;
  int ty = threadIdx.x >> 4;
  int c0 = blockIdx.x * 16;
  int r0 = blockIdx.y * 16;
  tile[ty][tx] = src[(size_t)(r0 + ty) * Cc + (c0 + tx)];
  __syncthreads();
  dst[(size_t)(c0 + ty) * R + (r0 + tx)] = f2bf(tile[tx][ty]);
}

// ---------------- GEMM: C[M,N] = A[M,K] * Bt[N,K]^T + bias -----------------
// Block (8 waves) computes a 128x64 C tile; all waves share one 64x32 B tile
// per K-step, staged in LDS (double-buffered, async fill when available).
// Each wave owns a 16x64 strip: 4 WMMAs per K-step reusing one A fragment.
#define BROW 40   /* padded LDS row stride (elems): conflict-free frag reads */

__global__ void __launch_bounds__(256)
k_gemm_wmma(const __bf16* __restrict__ A, const __bf16* __restrict__ Bt,
            const float* __restrict__ bias, float* __restrict__ outF,
            __bf16* __restrict__ outB, int N, int K, int storeBf16) {
  __shared__ alignas(16) __bf16 Bsh[2][64][BROW];

  int tid  = threadIdx.x;
  int lane = tid & 31;
  int w    = tid >> 5;            // wave in block: 0..7
  int half = lane >> 4;
  int n16  = lane & 15;
  int nBase = blockIdx.x * 64;
  int mRow  = blockIdx.y * 128 + w * 16 + n16;   // A row for this lane

  // cooperative B-tile fill mapping: 256 threads x 16B = 64 rows x 32 k
  int fn = tid >> 2;              // LDS/B row (n): 0..63
  int fs = (tid & 3) * 8;         // k segment start (elems): 0,8,16,24
  const __bf16* bsrc = Bt + (size_t)(nBase + fn) * K + fs;

  v8f acc[4];
#pragma unroll
  for (int j = 0; j < 4; ++j)
#pragma unroll
    for (int r = 0; r < 8; ++r) acc[j][r] = 0.f;

  // ---- prologue: fill LDS buffer 0, load first A fragment ----
#if USE_ASYNC_LDS
  __builtin_amdgcn_global_load_async_to_lds_b128(
      (gv4i*)bsrc, (lv4i*)&Bsh[0][fn][fs], 0, 0);
#else
  *(v8bf*)&Bsh[0][fn][fs] = *(const v8bf*)bsrc;
#endif

  const __bf16* arow = A + (size_t)mRow * K;
  v8bf a_lo = *(const v8bf*)(arow + half * 8);
  v8bf a_hi = *(const v8bf*)(arow + 16 + half * 8);

#if USE_ASYNC_LDS
  __builtin_amdgcn_s_wait_asynccnt(0);
#endif
  __syncthreads();

  int nK = K >> 5;                // K-steps of 32
  for (int kt = 0; kt < nK; ++kt) {
    int p = kt & 1;

    // ---- stage next K-step while computing this one ----
    v8bf a_nlo, a_nhi;
    if (kt + 1 < nK) {            // wave-uniform branch: EXEC stays full
      int kn = (kt + 1) << 5;
#if USE_ASYNC_LDS
      __builtin_amdgcn_global_load_async_to_lds_b128(
          (gv4i*)(bsrc + kn), (lv4i*)&Bsh[p ^ 1][fn][fs], 0, 0);
#else
      *(v8bf*)&Bsh[p ^ 1][fn][fs] = *(const v8bf*)(bsrc + kn);
#endif
      a_nlo = *(const v8bf*)(arow + kn + half * 8);
      a_nhi = *(const v8bf*)(arow + kn + 16 + half * 8);
    }

    // ---- compute: one A fragment, 4 B fragments from LDS, 4 WMMAs ----
    v16bf af;
#pragma unroll
    for (int e = 0; e < 8; ++e) { af[e] = a_lo[e]; af[8 + e] = a_hi[e]; }

#pragma unroll
    for (int j = 0; j < 4; ++j) {
      const v8bf* bp = (const v8bf*)&Bsh[p][j * 16 + n16][half * 8];
      v8bf blo = bp[0];
      v8bf bhi = bp[2];
      v16bf bfv;
#pragma unroll
      for (int e = 0; e < 8; ++e) { bfv[e] = blo[e]; bfv[8 + e] = bhi[e]; }
      acc[j] = __builtin_amdgcn_wmma_f32_16x16x32_bf16(
          false, af, false, bfv, (short)0, acc[j], false, false);
    }

    a_lo = a_nlo;
    a_hi = a_nhi;
#if USE_ASYNC_LDS
    __builtin_amdgcn_s_wait_asynccnt(0);
#endif
    __syncthreads();
  }

  // ---- epilogue: bias + store ----
  int mBase = blockIdx.y * 128 + w * 16;
#pragma unroll
  for (int j = 0; j < 4; ++j) {
    int ng = nBase + j * 16 + n16;
    float bv = bias[ng];
#pragma unroll
    for (int r = 0; r < 8; ++r) {
      int m = mBase + r + half * 8;
      float v = acc[j][r] + bv;
      if (storeBf16) outB[(size_t)m * N + ng] = f2bf(v);
      else           outF[(size_t)m * N + ng] = v;
    }
  }
}

// ---------------- Per-token attention over heads ---------------------------
// qkv row layout per token: col = h*192 + s*64 + d   (s: 0=q,1=k,2=v)
__global__ void __launch_bounds__(256)
k_attn_wmma(const __bf16* __restrict__ qkv, const unsigned char* __restrict__ mask,
            __bf16* __restrict__ outA) {
  __shared__ float plds[8][16][17];
  int lane = threadIdx.x & 31;
  int w    = threadIdx.x >> 5;
  int half = lane >> 4;
  int n16  = lane & 15;
  int t    = blockIdx.x * 8 + w;

  const __bf16* base = qkv + (size_t)t * (3 * C_SZ);

  // ---- S = Q @ K^T : M=h(16), N=g(16), K=d(64) -> two 16x16x32 WMMAs ----
  v8f s;
#pragma unroll
  for (int r = 0; r < 8; ++r) s[r] = 0.f;

#pragma unroll
  for (int kc = 0; kc < 2; ++kc) {
    const __bf16* qrow = base + n16 * 192;                 // Q row, head m=n16
    v8bf qlo = *(const v8bf*)(qrow + kc * 32 + half * 8);
    v8bf qhi = *(const v8bf*)(qrow + kc * 32 + 16 + half * 8);
    v16bf af;
#pragma unroll
    for (int e = 0; e < 8; ++e) { af[e] = qlo[e]; af[8 + e] = qhi[e]; }

    const __bf16* krow = base + n16 * 192 + 64 + kc * 32 + half * 8; // K row g=n16
    v8bf klo = *(const v8bf*)krow;
    v8bf khi = *(const v8bf*)(krow + 16);
    v16bf bfv;
#pragma unroll
    for (int e = 0; e < 8; ++e) { bfv[e] = klo[e]; bfv[8 + e] = khi[e]; }

    s = __builtin_amdgcn_wmma_f32_16x16x32_bf16(
        false, af, false, bfv, (short)0, s, false, false);
  }

  // ---- scale, mask(g), softmax over g (g = n = lane%16 -> lane butterflies)
  bool mk = mask[(size_t)t * H_SZ + n16] != 0;
  float p[8];
#pragma unroll
  for (int r = 0; r < 8; ++r) {
    float v = mk ? s[r] * 0.125f : -1e9f;
    float mx = v;
#pragma unroll
    for (int off = 1; off < 16; off <<= 1) mx = fmaxf(mx, __shfl_xor(mx, off, 32));
    float e = __expf(v - mx);
    float sum = e;
#pragma unroll
    for (int off = 1; off < 16; off <<= 1) sum += __shfl_xor(sum, off, 32);
    p[r] = e / sum;
  }

  // ---- re-layout P (C layout -> A fragment) via LDS ----
#pragma unroll
  for (int r = 0; r < 8; ++r) plds[w][r + half * 8][n16] = p[r];
  __syncthreads();

  __bf16 bz = __builtin_bit_cast(__bf16, (unsigned short)0);
  v16bf ap;
#pragma unroll
  for (int e = 0; e < 8; ++e) {
    ap[e]     = f2bf(plds[w][n16][half * 8 + e]);  // K = half*8+e (< 16)
    ap[8 + e] = bz;                                // K >= 16 padded with zero
  }

  // ---- O = P @ V : M=h(16), K=g(16 padded to 32), N=d(64) -> 4 WMMAs ----
#pragma unroll
  for (int jn = 0; jn < 4; ++jn) {
    v16bf bv;
#pragma unroll
    for (int e = 0; e < 16; ++e)
      bv[e] = (e < 8) ? base[(half * 8 + e) * 192 + 128 + jn * 16 + n16] : bz; // V[g=half*8+e][d]
    v8f o;
#pragma unroll
    for (int r = 0; r < 8; ++r) o[r] = 0.f;
    o = __builtin_amdgcn_wmma_f32_16x16x32_bf16(
        false, ap, false, bv, (short)0, o, false, false);
#pragma unroll
    for (int r = 0; r < 8; ++r)
      outA[(size_t)t * C_SZ + (r + half * 8) * HD_SZ + jn * 16 + n16] = f2bf(o[r]);
  }
}

// ---------------------------------------------------------------------------
extern "C" void kernel_launch(void* const* d_in, const int* in_sizes, int n_in,
                              void* d_out, int out_size, void* d_ws, size_t ws_size,
                              hipStream_t stream) {
  (void)in_sizes; (void)n_in; (void)out_size; (void)ws_size;
  const float* x            = (const float*)d_in[0];
  const unsigned char* mask = (const unsigned char*)d_in[1];
  const float* W_attn       = (const float*)d_in[2];
  const float* b_attn       = (const float*)d_in[3];
  const float* W_proj       = (const float*)d_in[4];
  const float* b_proj       = (const float*)d_in[5];
  float* out = (float*)d_out;

  // workspace layout (bytes)
  char* ws = (char*)d_ws;
  __bf16* xb  = (__bf16*)(ws);                                    // 32 MB
  __bf16* wat = (__bf16*)(ws + 33554432ull);                      // 6 MB
  __bf16* wpt = (__bf16*)(ws + 33554432ull + 6291456ull);         // 2 MB
  __bf16* qkv = (__bf16*)(ws + 33554432ull + 6291456ull + 2097152ull);          // 96 MB
  __bf16* att = (__bf16*)(ws + 33554432ull + 6291456ull + 2097152ull + 100663296ull); // 32 MB

  // 1) x -> bf16
  k_cvt_bf16<<<4096, 256, 0, stream>>>(x, xb, M_ROWS * K_SZ / 4);

  // 2) weights -> transposed bf16 (N x K)
  k_transpose_to_bf16<<<dim3(N1_SZ / 16, K_SZ / 16), 256, 0, stream>>>(W_attn, wat, K_SZ, N1_SZ);
  k_transpose_to_bf16<<<dim3(N2_SZ / 16, K_SZ / 16), 256, 0, stream>>>(W_proj, wpt, K_SZ, N2_SZ);

  // 3) qkv = x @ W_attn + b_attn  (bf16 out); grid: (N/64, M/128)
  k_gemm_wmma<<<dim3(N1_SZ / 64, M_ROWS / 128), 256, 0, stream>>>(
      xb, wat, b_attn, nullptr, qkv, N1_SZ, K_SZ, 1);

  // 4) per-token attention over heads
  k_attn_wmma<<<M_ROWS / 8, 256, 0, stream>>>(qkv, mask, att);

  // 5) out = att @ W_proj + b_proj  (f32 out)
  k_gemm_wmma<<<dim3(N2_SZ / 64, M_ROWS / 128), 256, 0, stream>>>(
      att, wpt, b_proj, out, nullptr, N2_SZ, K_SZ, 0);
}
